// GGNNSequential_48455821033927
// MI455X (gfx1250) — compile-verified, matrix operand order checked by reading, not run on hardware
//
#include <hip/hip_runtime.h>
#include <hip/hip_bf16.h>
#include <math.h>

// ---------------------------------------------------------------------------
// GGNN for MI455X (gfx1250, wave32).
// - GRU gate GEMMs + GRU nonlinearity fused into one WMMA kernel (6 gate
//   tiles per wave, 24 v_wmma per wave) -> no gi/gh round-trip (~7.4 GB saved).
// - Edge MLP layer-2 GEMM: 4 output tiles per wave (A-fragment reuse).
// - h/agg (51MB each) stay L2-resident (192MB L2); m streamed as bf16.
// ---------------------------------------------------------------------------

#define H      128
#define ND     64
#define ED     16
#define EPD    16
#define NSTATES 3
#define NSTEPS  4
#define CONDD  11
#define BOXE   8
#define MLPH   256

typedef __attribute__((ext_vector_type(16))) __bf16 v16bf;
typedef __attribute__((ext_vector_type(8)))  float  v8f;

__device__ __forceinline__ unsigned short f32_to_bf16(float f) {
    union { float f; unsigned int u; } x; x.f = f;
    unsigned int u = x.u;
    u += 0x7fffu + ((u >> 16) & 1u);           // round to nearest even
    return (unsigned short)(u >> 16);
}
__device__ __forceinline__ float bf16_to_f32(unsigned short h) {
    union { unsigned int u; float f; } x; x.u = ((unsigned int)h) << 16;
    return x.f;
}
__device__ __forceinline__ float sigmoidf_(float x) { return 1.0f / (1.0f + __expf(-x)); }

#define WMMA_BF16(a, b, c) \
    __builtin_amdgcn_wmma_f32_16x16x32_bf16(false, (a), false, (b), (short)0, (c), false, false)

// A-fragment packing address for 16-bit 16x32 A tile (ISA 7.12.2):
// lanes 0-15 hold K 0..7 & 16..23, lanes 16-31 hold K 8..15 & 24..31.
__device__ __forceinline__ long afrag_addr(long mt, int mm, int kt, int kk, int KT) {
    int lane = mm + (((kk >> 3) & 1) << 4);
    int half = (kk & 7) + (((kk >> 4) & 1) << 3);
    return ((mt * KT + kt) * 32 + lane) * 16 + half;
}

// ------------------------------ elementwise --------------------------------

__global__ void k_zero(float* __restrict__ p, long n) {
    long i = (long)blockIdx.x * blockDim.x + threadIdx.x;
    if (i < n) p[i] = 0.f;
}

// h = x @ in_W + in_b  ([N,64]@[64,128]); one block (128 thr) per node.
__global__ void k_node_in(const float* __restrict__ x, const float* __restrict__ W,
                          const float* __restrict__ b, float* __restrict__ h, int N) {
    int n = blockIdx.x, o = threadIdx.x;
    __shared__ float xr[ND];
    if (o < ND) xr[o] = x[(long)n * ND + o];
    __syncthreads();
    float acc = b[o];
#pragma unroll
    for (int c = 0; c < ND; ++c) acc += xr[c] * W[c * H + o];
    h[(long)n * H + o] = acc;
}

// ef = edge_attr @ ep_W + ep_b  ([E,16]@[16,16])
__global__ void k_edge_proj(const float* __restrict__ ea, const float* __restrict__ W,
                            const float* __restrict__ b, float* __restrict__ ef, long E) {
    long idx = (long)blockIdx.x * blockDim.x + threadIdx.x;
    if (idx >= E * EPD) return;
    long e = idx >> 4; int o = (int)(idx & 15);
    const float* row = ea + e * ED;
    float acc = b[o];
#pragma unroll
    for (int c = 0; c < ED; ++c) acc += row[c] * W[c * EPD + o];
    ef[idx] = acc;
}

// hidden = relu(ef @ msg_W1[s] + msg_b1[s]), written straight into bf16
// A-fragment layout (fused pack: saves one full [E,128] fp32 round-trip).
__global__ void k_edge_mlp1_pack(const float* __restrict__ ef, const float* __restrict__ W1,
                                 const float* __restrict__ b1,
                                 unsigned short* __restrict__ apack, long E) {
    long idx = (long)blockIdx.x * blockDim.x + threadIdx.x;
    if (idx >= E * (long)H) return;
    long e = idx >> 7; int k = (int)(idx & 127);
    const float* row = ef + e * EPD;
    float acc = b1[k];
#pragma unroll
    for (int c = 0; c < EPD; ++c) acc += row[c] * W1[c * H + k];
    acc = fmaxf(acc, 0.f);
    apack[afrag_addr(e >> 4, (int)(e & 15), k >> 5, k & 31, H / 32)] = f32_to_bf16(acc);
}

// fp32 row-major [M,128] -> bf16 A-fragment layout
__global__ void k_pack_a(const float* __restrict__ src, unsigned short* __restrict__ apack, long M) {
    long idx = (long)blockIdx.x * blockDim.x + threadIdx.x;
    if (idx >= M * (long)H) return;
    long n = idx >> 7; int k = (int)(idx & 127);
    apack[afrag_addr(n >> 4, (int)(n & 15), k >> 5, k & 31, H / 32)] = f32_to_bf16(src[idx]);
}

// Weight -> bf16 B-fragment layout.  Logical B[k,o]:
//   transpose=1: B[k,o] = W[o*K + k]   (for h @ W.T, W:[O,K])
//   transpose=0: B[k,o] = W[k*O + o]   (for h @ W,   W:[K,O])
__global__ void k_pack_b(const float* __restrict__ W, unsigned short* __restrict__ bpack,
                         int K, int O, int transpose) {
    int idx = blockIdx.x * blockDim.x + threadIdx.x;
    if (idx >= K * O) return;
    int k = idx / O, o = idx % O;
    float v = transpose ? W[o * K + k] : W[k * O + o];
    int lane = (o & 15) + (((k >> 3) & 1) << 4);
    int half = (k & 7) + (((k >> 4) & 1) << 3);
    bpack[(((long)(o >> 4) * (K / 32) + (k >> 5)) * 32 + lane) * 16 + half] = f32_to_bf16(v);
}

// ------------------------------ WMMA GEMM ----------------------------------
// One wave per 16 x (16*OTILE) output strip: each A fragment feeds OTILE
// WMMAs (A-operand reuse).  Tile selection is wave-uniform so EXEC stays
// all-ones around the WMMAs.
template <bool OUT_BF16, int OTILE>
__global__ void k_gemm_bf16(const unsigned short* __restrict__ Apack,
                            const unsigned short* __restrict__ Bpack,
                            const float* __restrict__ bias,
                            void* __restrict__ out,
                            long Mt, int Ot, int KT, int ldO) {
    int wave = threadIdx.x >> 5;
    int lane = threadIdx.x & 31;
    int og_n = Ot / OTILE;
    long tile = (long)blockIdx.x * (blockDim.x >> 5) + wave;
    if (tile >= Mt * (long)og_n) return;
    long mt = tile / og_n;
    int  og = (int)(tile % og_n);

    const v16bf* A = (const v16bf*)Apack;
    const v16bf* B = (const v16bf*)Bpack;
    v8f acc[OTILE];
#pragma unroll
    for (int oi = 0; oi < OTILE; ++oi) acc[oi] = (v8f){};
    for (int kt = 0; kt < KT; ++kt) {
        v16bf a = A[(mt * KT + kt) * 32 + lane];
#pragma unroll
        for (int oi = 0; oi < OTILE; ++oi) {
            v16bf b = B[((long)(og * OTILE + oi) * KT + kt) * 32 + lane];
            acc[oi] = WMMA_BF16(a, b, acc[oi]);
        }
    }
    // D layout: lane -> col n=lane&15; rows m = v + 8*(lane>=16)
    int n = lane & 15;
    int mbase = (lane >> 4) * 8;
#pragma unroll
    for (int oi = 0; oi < OTILE; ++oi) {
        int o = (og * OTILE + oi) * 16 + n;
        float bi = bias ? bias[o] : 0.f;
#pragma unroll
        for (int v = 0; v < 8; ++v) {
            long row = mt * 16 + mbase + v;
            float val = acc[oi][v] + bi;
            if (OUT_BF16) ((unsigned short*)out)[row * (long)ldO + o] = f32_to_bf16(val);
            else          ((float*)out)[row * (long)ldO + o] = val;
        }
    }
}

// --------------------- fused GRU gate GEMMs + update -----------------------
// One wave per (16-node, 16-feature) tile.  Computes all six gate tiles
// (ir,iz,inn from agg@Wih^T; hr,hz,hn from h@Whh^T) with 24 WMMAs, then
// applies the GRU nonlinearity on the D fragments in-register and writes h.
// Eliminates the [N,384] gi/gh fp32 round-trips entirely.
__global__ void k_gru_fused(const unsigned short* __restrict__ Aagg,
                            const unsigned short* __restrict__ Ah,
                            const unsigned short* __restrict__ Bih,   // Ot=24,KT=4
                            const unsigned short* __restrict__ Bhh,
                            const float* __restrict__ bih, const float* __restrict__ bhh,
                            float* __restrict__ h, long Mt) {
    int wave = threadIdx.x >> 5;
    int lane = threadIdx.x & 31;
    long tile = (long)blockIdx.x * (blockDim.x >> 5) + wave;
    if (tile >= Mt * 8) return;           // 8 feature tiles (H/16)
    long mt = tile >> 3;
    int  ot = (int)(tile & 7);
    const int KT = H / 32;                // 4

    const v16bf* AG = (const v16bf*)Aagg;
    const v16bf* AH = (const v16bf*)Ah;
    const v16bf* BI = (const v16bf*)Bih;
    const v16bf* BH = (const v16bf*)Bhh;

    v8f ir = {}, iz = {}, inn = {}, hr = {}, hz = {}, hn = {};
    for (int kt = 0; kt < KT; ++kt) {
        v16bf ag = AG[(mt * KT + kt) * 32 + lane];
        v16bf ah = AH[(mt * KT + kt) * 32 + lane];
        v16bf b0 = BI[((long)(ot     ) * KT + kt) * 32 + lane];
        v16bf b1 = BI[((long)(ot +  8) * KT + kt) * 32 + lane];
        v16bf b2 = BI[((long)(ot + 16) * KT + kt) * 32 + lane];
        v16bf c0 = BH[((long)(ot     ) * KT + kt) * 32 + lane];
        v16bf c1 = BH[((long)(ot +  8) * KT + kt) * 32 + lane];
        v16bf c2 = BH[((long)(ot + 16) * KT + kt) * 32 + lane];
        ir  = WMMA_BF16(ag, b0, ir);
        iz  = WMMA_BF16(ag, b1, iz);
        inn = WMMA_BF16(ag, b2, inn);
        hr  = WMMA_BF16(ah, c0, hr);
        hz  = WMMA_BF16(ah, c1, hz);
        hn  = WMMA_BF16(ah, c2, hn);
    }
    int n = lane & 15;
    int mbase = (lane >> 4) * 8;
    int o = ot * 16 + n;
    float bir = bih[o],       bhr = bhh[o];
    float biz = bih[H + o],   bhz = bhh[H + o];
    float bin = bih[2*H + o], bhn = bhh[2*H + o];
#pragma unroll
    for (int v = 0; v < 8; ++v) {
        long row = mt * 16 + mbase + v;
        float r  = sigmoidf_(ir[v] + bir + hr[v] + bhr);
        float z  = sigmoidf_(iz[v] + biz + hz[v] + bhz);
        float nn = tanhf(inn[v] + bin + r * (hn[v] + bhn));
        float ho = h[row * (long)H + o];
        h[row * (long)H + o] = (1.f - z) * nn + z * ho;
    }
}

// ------------------------- message pass scatter ----------------------------

// agg[i[e],k] += m[e,k] * h[j[e],k]   (h gathers + atomics hit L2: 51MB tables;
// m is the only HBM-streamed operand -> software prefetch 64K elements (128KB) ahead)
__global__ void k_msg_scatter(const unsigned short* __restrict__ m,
                              const float* __restrict__ h,
                              const int* __restrict__ ei,
                              float* __restrict__ agg, long E) {
    long idx = (long)blockIdx.x * blockDim.x + threadIdx.x;
    if (idx >= E * (long)H) return;
    long e = idx >> 7; int k = (int)(idx & 127);
    if (idx + 65536 < E * (long)H) __builtin_prefetch(&m[idx + 65536], 0, 0);
    int i = ei[e];
    int j = ei[E + e];
    float mv = bf16_to_f32(m[idx]);
    float hv = h[(long)j * H + k];
    atomicAdd(&agg[(long)i * H + k], mv * hv);
}

// LayerNorm over 128 feats; optional cond-gate (last state); writes the state
// buffer AND writes back into h (reference carries normalized h forward).
__global__ void k_layernorm(const float* __restrict__ hin, const float* __restrict__ g,
                            const float* __restrict__ b, float* __restrict__ sout,
                            float* __restrict__ hout,
                            const float* __restrict__ gate, const int* __restrict__ batch) {
    int n = blockIdx.x, k = threadIdx.x;
    __shared__ float red[H];
    float v = hin[(long)n * H + k];
    red[k] = v; __syncthreads();
    for (int s = H / 2; s > 0; s >>= 1) { if (k < s) red[k] += red[k + s]; __syncthreads(); }
    float mu = red[0] / H; __syncthreads();
    float d = v - mu;
    red[k] = d * d; __syncthreads();
    for (int s = H / 2; s > 0; s >>= 1) { if (k < s) red[k] += red[k + s]; __syncthreads(); }
    float var = red[0] / H;
    float out = d * rsqrtf(var + 1e-5f) * g[k] + b[k];
    if (gate) out *= gate[(long)batch[n] * H + k];
    sout[(long)n * H + k] = out;
    hout[(long)n * H + k] = out;
}

// -------------------------- cond / combine / head --------------------------

__global__ void k_cond(const float* __restrict__ ph, const float* __restrict__ te,
                       const float* __restrict__ rf, const int* __restrict__ box,
                       const float* __restrict__ bemb, float* __restrict__ cond, int B) {
    int idx = blockIdx.x * blockDim.x + threadIdx.x;
    if (idx >= B * CONDD) return;
    int b = idx / CONDD, c = idx % CONDD;
    float v;
    if (c == 0) v = ph[b];
    else if (c == 1) v = te[b];
    else if (c == 2) v = rf[b];
    else v = bemb[box[b] * BOXE + (c - 3)];
    cond[idx] = v;
}

__global__ void k_gate(const float* __restrict__ cond, const float* __restrict__ W,
                       const float* __restrict__ b, float* __restrict__ gate, int B) {
    int idx = blockIdx.x * blockDim.x + threadIdx.x;
    if (idx >= B * H) return;
    int bb = idx >> 7, o = idx & 127;
    float acc = b[o];
    for (int c = 0; c < CONDD; ++c) acc += cond[bb * CONDD + c] * W[c * H + o];
    gate[idx] = sigmoidf_(acc);
}

__global__ void k_hcond(const float* __restrict__ cond,
                        const float* __restrict__ W1, const float* __restrict__ b1,
                        const float* __restrict__ W2, const float* __restrict__ b2,
                        float* __restrict__ hcond) {
    int b = blockIdx.x, t = threadIdx.x;  // 64 threads
    __shared__ float t1[64];
    float acc = b1[t];
    for (int c = 0; c < CONDD; ++c) acc += cond[b * CONDD + c] * W1[c * 64 + t];
    t1[t] = fmaxf(acc, 0.f);
    __syncthreads();
    if (t < 32) {
        float a2 = b2[t];
        for (int k = 0; k < 64; ++k) a2 += t1[k] * W2[k * 32 + t];
        hcond[b * 32 + t] = a2;
    }
}

// Softmax-gated state combine + mean-pool accumulation (one block per node).
__global__ void k_combine(const float* __restrict__ s0, const float* __restrict__ s1,
                          const float* __restrict__ s2, const float* __restrict__ sgW,
                          const float* __restrict__ sgb, const int* __restrict__ batch,
                          float* __restrict__ sums, float* __restrict__ counts) {
    int n = blockIdx.x, k = threadIdx.x;   // 128 threads
    long off = (long)n * H + k;
    float v0 = s0[off], v1 = s1[off], v2 = s2[off];
    __shared__ float red[H];
    float logits[3];
    for (int s = 0; s < 3; ++s) {
        float p = v0 * sgW[k * 3 + s] + v1 * sgW[(H + k) * 3 + s] + v2 * sgW[(2 * H + k) * 3 + s];
        red[k] = p; __syncthreads();
        for (int r = H / 2; r > 0; r >>= 1) { if (k < r) red[k] += red[k + r]; __syncthreads(); }
        logits[s] = red[0] + sgb[s]; __syncthreads();
    }
    float mx = fmaxf(logits[0], fmaxf(logits[1], logits[2]));
    float e0 = __expf(logits[0] - mx), e1 = __expf(logits[1] - mx), e2 = __expf(logits[2] - mx);
    float inv = 1.f / (e0 + e1 + e2);
    float comb = (v0 * e0 + v1 * e1 + v2 * e2) * inv;
    int b = batch[n];
    atomicAdd(&sums[(long)b * H + k], comb);
    if (k == 0) atomicAdd(&counts[b], 1.0f);
}

// Final per-graph MLP head (one block of 256 threads per graph).
__global__ void k_final(const float* __restrict__ sums, const float* __restrict__ counts,
                        const float* __restrict__ hcond,
                        const float* __restrict__ W1, const float* __restrict__ b1,
                        const float* __restrict__ W2, const float* __restrict__ b2,
                        const float* __restrict__ W3, const float* __restrict__ b3,
                        float* __restrict__ out) {
    int b = blockIdx.x, t = threadIdx.x;   // 256 threads
    __shared__ float vec[H + 32];
    __shared__ float z1[MLPH];
    __shared__ float z2[128];
    __shared__ float red[128];
    float cnt = fmaxf(counts[b], 1.f);
    if (t < H) vec[t] = sums[(long)b * H + t] / cnt;
    else if (t < H + 32) vec[t] = hcond[b * 32 + (t - H)];
    __syncthreads();
    {
        float acc = b1[t];
        for (int c = 0; c < H + 32; ++c) acc += vec[c] * W1[c * MLPH + t];
        z1[t] = fmaxf(acc, 0.f);
    }
    __syncthreads();
    if (t < 128) {
        float acc = b2[t];
        for (int c = 0; c < MLPH; ++c) acc += z1[c] * W2[c * 128 + t];
        z2[t] = fmaxf(acc, 0.f);
    }
    __syncthreads();
    if (t < 128) red[t] = z2[t] * W3[t];
    __syncthreads();
    for (int s = 64; s > 0; s >>= 1) { if (t < s && t + s < 128) red[t] += red[t + s]; __syncthreads(); }
    if (t == 0) out[b] = red[0] + b3[0];
}

// ------------------------------- launch ------------------------------------

extern "C" void kernel_launch(void* const* d_in, const int* in_sizes, int n_in,
                              void* d_out, int out_size, void* d_ws, size_t ws_size,
                              hipStream_t stream) {
    (void)n_in; (void)out_size; (void)ws_size;
    // inputs (setup_inputs dict order; params flattened in make_params order)
    const float* x      = (const float*)d_in[0];
    const int*   ei     = (const int*)  d_in[1];
    const float* ea     = (const float*)d_in[2];
    const int*   batch  = (const int*)  d_in[3];
    const float* ph     = (const float*)d_in[4];
    const float* te     = (const float*)d_in[5];
    const float* rf     = (const float*)d_in[6];
    const int*   box    = (const int*)  d_in[7];
    const float* in_W   = (const float*)d_in[8];
    const float* in_b   = (const float*)d_in[9];
    const float* ep_W   = (const float*)d_in[10];
    const float* ep_b   = (const float*)d_in[11];
    const float* msg_W1 = (const float*)d_in[12];
    const float* msg_b1 = (const float*)d_in[13];
    const float* msg_W2 = (const float*)d_in[14];
    const float* msg_b2 = (const float*)d_in[15];
    const float* gWih   = (const float*)d_in[16];
    const float* gWhh   = (const float*)d_in[17];
    const float* gbih   = (const float*)d_in[18];
    const float* gbhh   = (const float*)d_in[19];
    const float* ln_g   = (const float*)d_in[20];
    const float* ln_b   = (const float*)d_in[21];
    const float* cg_W   = (const float*)d_in[22];
    const float* cg_b   = (const float*)d_in[23];
    const float* sg_W   = (const float*)d_in[24];
    const float* sg_b   = (const float*)d_in[25];
    const float* bemb   = (const float*)d_in[26];
    const float* cp_W1  = (const float*)d_in[27];
    const float* cp_b1  = (const float*)d_in[28];
    const float* cp_W2  = (const float*)d_in[29];
    const float* cp_b2  = (const float*)d_in[30];
    const float* m_W1   = (const float*)d_in[31];
    const float* m_b1   = (const float*)d_in[32];
    const float* m_W2   = (const float*)d_in[33];
    const float* m_b2   = (const float*)d_in[34];
    const float* m_W3   = (const float*)d_in[35];
    const float* m_b3   = (const float*)d_in[36];

    const int  N = in_sizes[0] / ND;    // 100000 (multiple of 16)
    const long E = in_sizes[1] / 2;     // 640000 (multiple of 16)
    const int  B = in_sizes[4];         // 256
    float* out = (float*)d_out;

    // carve workspace
    char* ws = (char*)d_ws; size_t off = 0;
    auto carve = [&](size_t bytes) -> void* {
        void* p = ws + off; off = (off + bytes + 255) & ~(size_t)255; return p;
    };
    float* h_buf  = (float*)carve((size_t)N * H * 4);
    float* agg    = (float*)carve((size_t)N * H * 4);
    float* ef     = (float*)carve((size_t)E * EPD * 4);
    unsigned short* mbuf   = (unsigned short*)carve((size_t)E * H * 2);
    unsigned short* ehpack = (unsigned short*)carve((size_t)E * H * 2);
    unsigned short* apack  = (unsigned short*)carve((size_t)N * H * 2);
    unsigned short* hpack  = (unsigned short*)carve((size_t)N * H * 2);
    float* sbuf[NSTATES];
    for (int s = 0; s < NSTATES; ++s) sbuf[s] = (float*)carve((size_t)N * H * 4);
    float* cond   = (float*)carve((size_t)B * CONDD * 4);
    float* gate   = (float*)carve((size_t)B * H * 4);
    float* hcond  = (float*)carve((size_t)B * 32 * 4);
    float* sums   = (float*)carve((size_t)B * H * 4);
    float* counts = (float*)carve((size_t)B * 4);
    unsigned short* bih_p = (unsigned short*)carve((size_t)3 * H * H * 2);
    unsigned short* bhh_p = (unsigned short*)carve((size_t)3 * H * H * 2);
    unsigned short* w2_p  = (unsigned short*)carve((size_t)H * H * 2);

    const int T = 256;
    auto cdiv = [](long a, long b) { return (int)((a + b - 1) / b); };

    // ---- setup ----
    k_node_in<<<N, H, 0, stream>>>(x, in_W, in_b, h_buf, N);
    k_edge_proj<<<cdiv(E * EPD, T), T, 0, stream>>>(ea, ep_W, ep_b, ef, E);
    k_cond<<<cdiv((long)B * CONDD, T), T, 0, stream>>>(ph, te, rf, box, bemb, cond, B);
    k_gate<<<cdiv((long)B * H, T), T, 0, stream>>>(cond, cg_W, cg_b, gate, B);
    k_hcond<<<B, 64, 0, stream>>>(cond, cp_W1, cp_b1, cp_W2, cp_b2, hcond);
    k_zero<<<cdiv((long)B * H, T), T, 0, stream>>>(sums, (long)B * H);
    k_zero<<<cdiv((long)B, T), T, 0, stream>>>(counts, B);

    const long edgeTiles = (E / 16) * ((H / 16) / 4);     // OTILE=4 strips
    const long gruTiles  = (long)(N / 16) * (H / 16);     // fused GRU tiles

    for (int s = 0; s < NSTATES; ++s) {
        // pack per-state weights into B fragments
        k_pack_b<<<cdiv(3 * H * H, T), T, 0, stream>>>(gWih + (size_t)s * 3 * H * H, bih_p, H, 3 * H, 1);
        k_pack_b<<<cdiv(3 * H * H, T), T, 0, stream>>>(gWhh + (size_t)s * 3 * H * H, bhh_p, H, 3 * H, 1);
        k_pack_b<<<cdiv(H * H, T), T, 0, stream>>>(msg_W2 + (size_t)s * H * H, w2_p, H, H, 0);

        // edge MLP: layer1 fused with A-fragment packing, layer2 on WMMA -> m (bf16)
        k_edge_mlp1_pack<<<cdiv(E * H, T), T, 0, stream>>>(
            ef, msg_W1 + (size_t)s * EPD * H, msg_b1 + (size_t)s * H, ehpack, E);
        k_gemm_bf16<true, 4><<<cdiv(edgeTiles, 8), 256, 0, stream>>>(
            ehpack, w2_p, msg_b2 + (size_t)s * H, mbuf, E / 16, H / 16, H / 32, H);

        for (int step = 0; step < NSTEPS; ++step) {
            k_zero<<<cdiv((long)N * H, T), T, 0, stream>>>(agg, (long)N * H);
            k_msg_scatter<<<cdiv(E * H, T), T, 0, stream>>>(mbuf, h_buf, ei, agg, E);
            k_pack_a<<<cdiv((long)N * H, T), T, 0, stream>>>(agg, apack, N);
            k_pack_a<<<cdiv((long)N * H, T), T, 0, stream>>>(h_buf, hpack, N);
            k_gru_fused<<<cdiv(gruTiles, 8), 256, 0, stream>>>(
                apack, hpack, bih_p, bhh_p,
                gbih + (size_t)s * 3 * H, gbhh + (size_t)s * 3 * H, h_buf, N / 16);
        }
        k_layernorm<<<N, H, 0, stream>>>(h_buf, ln_g + (size_t)s * H, ln_b + (size_t)s * H,
                                         sbuf[s], h_buf,
                                         (s == NSTATES - 1) ? gate : nullptr, batch);
    }

    k_combine<<<N, H, 0, stream>>>(sbuf[0], sbuf[1], sbuf[2], sg_W, sg_b, batch, sums, counts);
    k_final<<<B, 256, 0, stream>>>(sums, counts, hcond, m_W1, m_b1, m_W2, m_b2, m_W3, m_b3, out);
}
